// GraspEvaluator_2680059593186
// MI455X (gfx1250) — matrix operand order repaired and checked by reading, not compiled, not run on hardware
//
#include <hip/hip_runtime.h>
#include <hip/hip_bf16.h>

// ---------------------------------------------------------------------------
// GraspEvaluator forward for MI455X (gfx1250, wave32).
// All MLP/GEMM math on v_wmma_f32_16x16x32_f16.  Activation tiles in LDS are
// point-major so B fragments are contiguous (ds_load_b128 x2) and D fragments
// are contiguous (ds_store_b128 x1).  BN affine precomputed per layer in LDS.
// BatchNorm couples the whole batch -> 4-pass recompute per SA stage.
// ---------------------------------------------------------------------------

typedef __attribute__((ext_vector_type(16))) _Float16 v16h;
typedef __attribute__((ext_vector_type(8)))  _Float16 v8h;
typedef __attribute__((ext_vector_type(8)))  float    v8f;

#define BB     128
#define NPTS1  4224      // 4096 + 128
#define S1     128
#define NS1    64
#define S2     32
#define NS2    128
#define EPSB   1e-5f

// ---------------------------------------------------------------------------
// WMMA tile: D = W[row0..+15][0..K) x X^T tile.
//  W: f16 global, row-major [Co][ldw]  (A fragment: two contiguous 8-half runs)
//  X: f16 LDS, point-major  [pt][ldx]  (B fragment: two contiguous 8-half runs)
// wave32 layouts per CDNA5 ISA 7.12.2.
// ---------------------------------------------------------------------------
__device__ __forceinline__ v8f wmma_f16_t(const _Float16* __restrict__ W, int ldw, int row0,
                                          const _Float16* X, int ldx, int col0, int K)
{
  const int lane = threadIdx.x & 31;
  const int n  = lane & 15;
  const int kh = lane >> 4;
  const int m  = row0 + n;
  v8f acc = {};
  for (int k0 = 0; k0 < K; k0 += 32) {
    v8h a0 = *(const v8h*)(W + m * ldw + k0 + 8 * kh);
    v8h a1 = *(const v8h*)(W + m * ldw + k0 + 16 + 8 * kh);
    v8h b0 = *(const v8h*)(X + (col0 + n) * ldx + k0 + 8 * kh);
    v8h b1 = *(const v8h*)(X + (col0 + n) * ldx + k0 + 16 + 8 * kh);
    v16h a  = __builtin_shufflevector(a0, a1, 0,1,2,3,4,5,6,7,8,9,10,11,12,13,14,15);
    v16h bm = __builtin_shufflevector(b0, b1, 0,1,2,3,4,5,6,7,8,9,10,11,12,13,14,15);
    acc = __builtin_amdgcn_wmma_f32_16x16x32_f16(false, a, false, bm, (short)0, acc,
                                                 false, false);
  }
  return acc;
}

// acc[v] = D[row0 + 8*kh + v][col0 + n]
__device__ __forceinline__ void stats_acc(v8f acc, int row0, float* sred)
{
  const int kh = (threadIdx.x >> 4) & 1;
#pragma unroll
  for (int v = 0; v < 8; ++v) {
    int ch = row0 + 8 * kh + v;
    float x = acc[v];
    atomicAdd(&sred[ch * 2], x);
    atomicAdd(&sred[ch * 2 + 1], x * x);
  }
}

// Precompute BN affine: out[2ch]=rsqrt(var+eps), out[2ch+1]=-mean*scale
__device__ __forceinline__ void bn_make(const float* st, float invM, float* out, int C)
{
  for (int ch = threadIdx.x; ch < C; ch += blockDim.x) {
    float mean = st[ch * 2] * invM;
    float var  = st[ch * 2 + 1] * invM - mean * mean;
    float inv  = rsqrtf(var + EPSB);
    out[ch * 2]     = inv;
    out[ch * 2 + 1] = -mean * inv;
  }
}

__device__ __forceinline__ void bn_relu_store_t(v8f acc, int row0, int col0,
                                                const float* bn, _Float16* Y, int ldy)
{
  const int lane = threadIdx.x & 31;
  const int n  = lane & 15;
  const int kh = lane >> 4;
  v8h y;
#pragma unroll
  for (int v = 0; v < 8; ++v) {
    int ch = row0 + 8 * kh + v;
    y[v] = (_Float16)fmaxf(fmaf(acc[v], bn[ch * 2], bn[ch * 2 + 1]), 0.f);
  }
  *(v8h*)(Y + (col0 + n) * ldy + row0 + 8 * kh) = y;
}

__device__ __forceinline__ void bn_relu_max(v8f acc, int row0, const float* bn, int* mx)
{
  const int kh = (threadIdx.x >> 4) & 1;
#pragma unroll
  for (int v = 0; v < 8; ++v) {
    int ch = row0 + 8 * kh + v;
    float y = fmaxf(fmaf(acc[v], bn[ch * 2], bn[ch * 2 + 1]), 0.f);
    atomicMax(&mx[ch], __float_as_int(y));   // nonneg floats: int order == float order
  }
}

__device__ __forceinline__ void flush_stats(const float* sred, float* gst, int nfloats)
{
  for (int t = threadIdx.x; t < nfloats; t += blockDim.x)
    atomicAdd(&gst[t], sred[t]);
}

// ---------------------------------------------------------------------------
// Misc small kernels
// ---------------------------------------------------------------------------
__global__ void concat_xyz_kernel(const float* __restrict__ pc,
                                  const float* __restrict__ gpc,
                                  float* __restrict__ xyz)
{
  int t = blockIdx.x * blockDim.x + threadIdx.x;
  if (t >= BB * NPTS1 * 3) return;
  int c = t % 3;
  int bi = t / 3;
  int b = bi / NPTS1;
  int i = bi % NPTS1;
  xyz[t] = (i < 4096) ? pc[((size_t)b * 4096 + i) * 3 + c]
                      : gpc[((size_t)b * 128 + (i - 4096)) * 3 + c];
}

__global__ void cvt_w_kernel(const float* __restrict__ src, _Float16* __restrict__ dst,
                             int Co, int Ci, int Cip)
{
  int t = blockIdx.x * blockDim.x + threadIdx.x;
  if (t >= Co * Cip) return;
  int r = t / Cip, c = t % Cip;
  dst[t] = (c < Ci) ? (_Float16)src[r * Ci + c] : (_Float16)0.f;
}

// FPS: one block per batch, dists in registers, argmax with first-index tiebreak.
__global__ __launch_bounds__(256) void fps_kernel(const float* __restrict__ xyz,
                                                  int N, int npoint,
                                                  int* __restrict__ fidx)
{
  const int b = blockIdx.x, tid = threadIdx.x;
  const float* X = xyz + (size_t)b * N * 3;
  float dist[17];
#pragma unroll
  for (int t = 0; t < 17; ++t) dist[t] = 1e10f;
  __shared__ float rv[256];
  __shared__ int   ri[256];
  int far = 0;
  for (int it = 0; it < npoint; ++it) {
    if (tid == 0) fidx[b * npoint + it] = far;
    float cx = X[far * 3], cy = X[far * 3 + 1], cz = X[far * 3 + 2];
    float best = -1.f; int bi = 0;
#pragma unroll
    for (int t = 0; t < 17; ++t) {
      int i = tid + t * 256;
      if (i < N) {
        float dx = X[i * 3] - cx, dy = X[i * 3 + 1] - cy, dz = X[i * 3 + 2] - cz;
        float d = dx * dx + dy * dy + dz * dz;
        float nd = fminf(dist[t], d);
        dist[t] = nd;
        if (nd > best) { best = nd; bi = i; }
      }
    }
    rv[tid] = best; ri[tid] = bi;
    __syncthreads();
    for (int s = 128; s > 0; s >>= 1) {
      if (tid < s) {
        if (rv[tid + s] > rv[tid] ||
            (rv[tid + s] == rv[tid] && ri[tid + s] < ri[tid])) {
          rv[tid] = rv[tid + s]; ri[tid] = ri[tid + s];
        }
      }
      __syncthreads();
    }
    far = ri[0];
    __syncthreads();
  }
}

__global__ void gather_nx_kernel(const float* __restrict__ xyz, const int* __restrict__ fidx,
                                 float* __restrict__ nx, int N, int S)
{
  int t = blockIdx.x * blockDim.x + threadIdx.x;
  if (t >= BB * S * 3) return;
  int c = t % 3;
  int bs = t / 3;
  int b = bs / S;
  nx[t] = xyz[((size_t)b * N + fidx[bs]) * 3 + c];
}

// Ball query: first `ns` hits in index order, pad with first hit.
__global__ void ballq_kernel(const float* __restrict__ xyz, const float* __restrict__ nxz,
                             int N, int S, int ns, float r2, int* __restrict__ gidx)
{
  int t = blockIdx.x * blockDim.x + threadIdx.x;
  if (t >= BB * S) return;
  int b = t / S;
  const float* X = xyz + (size_t)b * N * 3;
  float cx = nxz[t * 3], cy = nxz[t * 3 + 1], cz = nxz[t * 3 + 2];
  int* out = gidx + (size_t)t * ns;
  int cnt = 0, first = 0;
  for (int i = 0; i < N && cnt < ns; ++i) {
    float dx = X[i * 3] - cx, dy = X[i * 3 + 1] - cy, dz = X[i * 3 + 2] - cz;
    if (dx * dx + dy * dy + dz * dz < r2) {
      if (cnt == 0) first = i;
      out[cnt++] = i;
    }
  }
  for (; cnt < ns; ++cnt) out[cnt] = first;
}

// ---------------------------------------------------------------------------
// SA stage 1: one block per (b,s) group; 4 waves; 7(->32 pad)->64->64->128, n=64.
// LDS tiles point-major.  aux: bn0@0(128) bn1@128(128) bn2@256(256);
// stats scratch aliases the measured layer's bn slot.
// ---------------------------------------------------------------------------
__global__ __launch_bounds__(128) void sa1_pass_kernel(
    const float* __restrict__ xyz, const float* __restrict__ nx1,
    const int* __restrict__ gidx1,
    const _Float16* __restrict__ w0, const _Float16* __restrict__ w1,
    const _Float16* __restrict__ w2,
    float* __restrict__ st, float* __restrict__ feats1, int pass)
{
  __shared__ __align__(16) _Float16 X0[64][32];
  __shared__ __align__(16) _Float16 Y1[64][64];
  __shared__ __align__(16) _Float16 Y2[64][64];
  __shared__ float aux[512];
  __shared__ int   mx[128];
  __shared__ int   idxbuf[64];

  const int tid  = threadIdx.x;
  const int wave = tid >> 5;
  const int b    = blockIdx.x >> 7;
  const int s    = blockIdx.x & 127;
  const float invM = 1.f / (float)(BB * S1 * NS1);

  const int sredoff = (pass == 0) ? 0 : (pass == 1) ? 128 : 256;
  float* sred = aux + sredoff;
  const int nst = (pass == 2) ? 256 : 128;

  if (pass < 3) for (int t = tid; t < nst; t += 128) sred[t] = 0.f;
  else if (tid < 128) mx[tid] = 0;
  if (pass >= 1) bn_make(st,       invM, aux,       64);
  if (pass >= 2) bn_make(st + 128, invM, aux + 128, 64);
  if (pass == 3) bn_make(st + 256, invM, aux + 256, 128);
  if (tid < 64) idxbuf[tid] = gidx1[(b * S1 + s) * NS1 + tid];
  __syncthreads();

  for (int t = tid; t < 64 * 32; t += 128) {   // point-major gather
    int j = t >> 5, ch = t & 31;
    int i = idxbuf[j];
    float v = 0.f;
    if (ch < 3)       v = xyz[((size_t)b * NPTS1 + i) * 3 + ch] - nx1[(b * S1 + s) * 3 + ch];
    else if (ch < 6)  v = xyz[((size_t)b * NPTS1 + i) * 3 + (ch - 3)];
    else if (ch == 6) v = (i < 4096) ? 1.f : 0.f;
    X0[j][ch] = (_Float16)v;
  }
  __syncthreads();

  for (int ct = 0; ct < 4; ++ct) {
    v8f acc = wmma_f16_t(w0, 32, wave * 16, &X0[0][0], 32, ct * 16, 32);
    if (pass == 0) stats_acc(acc, wave * 16, sred);
    else           bn_relu_store_t(acc, wave * 16, ct * 16, aux, &Y1[0][0], 64);
  }
  __syncthreads();
  if (pass == 0) { flush_stats(sred, st, 128); return; }

  for (int ct = 0; ct < 4; ++ct) {
    v8f acc = wmma_f16_t(w1, 64, wave * 16, &Y1[0][0], 64, ct * 16, 64);
    if (pass == 1) stats_acc(acc, wave * 16, sred);
    else           bn_relu_store_t(acc, wave * 16, ct * 16, aux + 128, &Y2[0][0], 64);
  }
  __syncthreads();
  if (pass == 1) { flush_stats(sred, st + 128, 128); return; }

  for (int rr = 0; rr < 2; ++rr) {
    int rt = wave + rr * 4;
    for (int ct = 0; ct < 4; ++ct) {
      v8f acc = wmma_f16_t(w2, 64, rt * 16, &Y2[0][0], 64, ct * 16, 64);
      if (pass == 2) stats_acc(acc, rt * 16, sred);
      else           bn_relu_max(acc, rt * 16, aux + 256, mx);
    }
  }
  __syncthreads();
  if (pass == 2) flush_stats(sred, st + 256, 256);
  else if (tid < 128)
    feats1[((size_t)b * 128 + tid) * S1 + s] = __int_as_float(mx[tid]);
}

// ---------------------------------------------------------------------------
// SA stage 2: one block per (b,s); 8 waves; 131(->160)->128->128->256,
// n=128 in two 64-point chunks.  aux: bn0@0(256) bn1@256(256) bn2@512(512).
// ---------------------------------------------------------------------------
__global__ __launch_bounds__(256) void sa2_pass_kernel(
    const float* __restrict__ nx1, const float* __restrict__ feats1,
    const float* __restrict__ nx2, const int* __restrict__ gidx2,
    const _Float16* __restrict__ w0, const _Float16* __restrict__ w1,
    const _Float16* __restrict__ w2,
    float* __restrict__ st, float* __restrict__ feats2, int pass)
{
  __shared__ __align__(16) _Float16 X0[64][160];
  __shared__ __align__(16) _Float16 Y1[64][128];
  __shared__ __align__(16) _Float16 Y2[64][128];
  __shared__ float aux[1024];
  __shared__ int   mx[256];
  __shared__ int   idxbuf[64];

  const int tid  = threadIdx.x;
  const int wave = tid >> 5;
  const int b    = blockIdx.x >> 5;
  const int s    = blockIdx.x & 31;
  const float invM = 1.f / (float)(BB * S2 * NS2);

  const int sredoff = (pass == 0) ? 0 : (pass == 1) ? 256 : 512;
  float* sred = aux + sredoff;
  const int nst = (pass == 2) ? 512 : 256;

  if (pass < 3) for (int t = tid; t < nst; t += 256) sred[t] = 0.f;
  else if (tid < 256) mx[tid] = 0;
  if (pass >= 1) bn_make(st,       invM, aux,       128);
  if (pass >= 2) bn_make(st + 256, invM, aux + 256, 128);
  if (pass == 3) bn_make(st + 512, invM, aux + 512, 256);

  for (int chunk = 0; chunk < 2; ++chunk) {
    __syncthreads();
    if (tid < 64) idxbuf[tid] = gidx2[(b * S2 + s) * NS2 + chunk * 64 + tid];
    __syncthreads();
    for (int t = tid; t < 64 * 160; t += 256) {   // point-major gather
      int j = t / 160, ch = t % 160;
      int i = idxbuf[j];
      float v = 0.f;
      if (ch < 3)        v = nx1[(b * S1 + i) * 3 + ch] - nx2[(b * S2 + s) * 3 + ch];
      else if (ch < 131) v = feats1[((size_t)b * 128 + (ch - 3)) * S1 + i];
      X0[j][ch] = (_Float16)v;
    }
    __syncthreads();

    for (int ct = 0; ct < 4; ++ct) {
      v8f acc = wmma_f16_t(w0, 160, wave * 16, &X0[0][0], 160, ct * 16, 160);
      if (pass == 0) stats_acc(acc, wave * 16, sred);
      else           bn_relu_store_t(acc, wave * 16, ct * 16, aux, &Y1[0][0], 128);
    }
    if (pass == 0) continue;
    __syncthreads();

    for (int ct = 0; ct < 4; ++ct) {
      v8f acc = wmma_f16_t(w1, 128, wave * 16, &Y1[0][0], 128, ct * 16, 128);
      if (pass == 1) stats_acc(acc, wave * 16, sred);
      else           bn_relu_store_t(acc, wave * 16, ct * 16, aux + 256, &Y2[0][0], 128);
    }
    if (pass == 1) continue;
    __syncthreads();

    for (int rr = 0; rr < 2; ++rr) {
      int rt = wave + rr * 8;
      for (int ct = 0; ct < 4; ++ct) {
        v8f acc = wmma_f16_t(w2, 128, rt * 16, &Y2[0][0], 128, ct * 16, 128);
        if (pass == 2) stats_acc(acc, rt * 16, sred);
        else           bn_relu_max(acc, rt * 16, aux + 512, mx);
      }
    }
  }
  __syncthreads();
  if      (pass == 0) flush_stats(sred, st,       256);
  else if (pass == 1) flush_stats(sred, st + 256, 256);
  else if (pass == 2) flush_stats(sred, st + 512, 512);
  else if (tid < 256)
    feats2[((size_t)b * 256 + tid) * S2 + s] = __int_as_float(mx[tid]);
}

// ---------------------------------------------------------------------------
// SA stage 3 (global): one block per batch; 259(->288)->256->256->512, 32 pts.
// aux: bn0@0(512) bn1@512(512) bn2@1024(1024); sred aliases measured slot.
// ---------------------------------------------------------------------------
__global__ __launch_bounds__(256) void sa3_pass_kernel(
    const float* __restrict__ nx2, const float* __restrict__ feats2,
    const _Float16* __restrict__ w0, const _Float16* __restrict__ w1,
    const _Float16* __restrict__ w2,
    float* __restrict__ st, float* __restrict__ sa3o, int pass)
{
  __shared__ __align__(16) _Float16 X0[32][288];
  __shared__ __align__(16) _Float16 Y1[32][256];
  __shared__ __align__(16) _Float16 Y2[32][256];
  __shared__ float aux[2048];
  __shared__ int   mx[512];

  const int tid = threadIdx.x, wave = tid >> 5;
  const int b = blockIdx.x;
  const float invM = 1.f / (float)(BB * S2);

  const int sredoff = (pass == 0) ? 0 : (pass == 1) ? 512 : 1024;
  float* sred = aux + sredoff;
  const int nst = (pass == 2) ? 1024 : 512;

  if (pass < 3) for (int t = tid; t < nst; t += 256) sred[t] = 0.f;
  else          for (int t = tid; t < 512; t += 256) mx[t] = 0;
  if (pass >= 1) bn_make(st,        invM, aux,        256);
  if (pass >= 2) bn_make(st + 512,  invM, aux + 512,  256);
  if (pass == 3) bn_make(st + 1024, invM, aux + 1024, 512);

  for (int t = tid; t < 32 * 288; t += 256) {   // point-major gather
    int j = t / 288, ch = t % 288;
    float v = 0.f;
    if (ch < 3)        v = nx2[(b * S2 + j) * 3 + ch];
    else if (ch < 259) v = feats2[((size_t)b * 256 + (ch - 3)) * S2 + j];
    X0[j][ch] = (_Float16)v;
  }
  __syncthreads();

  for (int rr = 0; rr < 2; ++rr) {
    int rt = wave + rr * 8;
    for (int ct = 0; ct < 2; ++ct) {
      v8f acc = wmma_f16_t(w0, 288, rt * 16, &X0[0][0], 288, ct * 16, 288);
      if (pass == 0) stats_acc(acc, rt * 16, sred);
      else           bn_relu_store_t(acc, rt * 16, ct * 16, aux, &Y1[0][0], 256);
    }
  }
  __syncthreads();
  if (pass == 0) { flush_stats(sred, st, 512); return; }

  for (int rr = 0; rr < 2; ++rr) {
    int rt = wave + rr * 8;
    for (int ct = 0; ct < 2; ++ct) {
      v8f acc = wmma_f16_t(w1, 256, rt * 16, &Y1[0][0], 256, ct * 16, 256);
      if (pass == 1) stats_acc(acc, rt * 16, sred);
      else           bn_relu_store_t(acc, rt * 16, ct * 16, aux + 512, &Y2[0][0], 256);
    }
  }
  __syncthreads();
  if (pass == 1) { flush_stats(sred, st + 512, 512); return; }

  for (int rr = 0; rr < 4; ++rr) {
    int rt = wave + rr * 8;
    for (int ct = 0; ct < 2; ++ct) {
      v8f acc = wmma_f16_t(w2, 256, rt * 16, &Y2[0][0], 256, ct * 16, 256);
      if (pass == 2) stats_acc(acc, rt * 16, sred);
      else           bn_relu_max(acc, rt * 16, aux + 1024, mx);
    }
  }
  __syncthreads();
  if (pass == 2) flush_stats(sred, st + 1024, 1024);
  else for (int t = tid; t < 512; t += 256)
    sa3o[(size_t)b * 512 + t] = __int_as_float(mx[t]);
}

// ---------------------------------------------------------------------------
// FC GEMM: Y[b][o] = sum_k X[b][k] * W[o][k] + bias[o].  One wave per 16x16 tile.
// W is f16 [N][K] row-major -> contiguous B fragments.
// ---------------------------------------------------------------------------
__global__ __launch_bounds__(256) void fc_gemm_kernel(
    const float* __restrict__ X, const _Float16* __restrict__ W,
    const float* __restrict__ bias, float* __restrict__ Y, int K, int N)
{
  int gwid = (blockIdx.x * 256 + threadIdx.x) >> 5;
  int ntN = N >> 4;
  int rt = gwid & 7;
  int nt = gwid >> 3;
  if (nt >= ntN) return;
  const int lane = threadIdx.x & 31;
  const int n  = lane & 15;
  const int kh = lane >> 4;
  const int row0 = rt * 16, col0 = nt * 16;
  const int m = row0 + n;
  v8f acc = {};
  for (int k0 = 0; k0 < K; k0 += 32) {
    v16h a;
#pragma unroll
    for (int v = 0; v < 8; ++v) {
      int ka = (v < 4) ? (k0 + 8 * kh + 2 * v) : (k0 + 16 + 8 * kh + 2 * (v - 4));
      a[2 * v]     = (_Float16)X[m * K + ka];
      a[2 * v + 1] = (_Float16)X[m * K + ka + 1];
    }
    v8h b0 = *(const v8h*)(W + (col0 + n) * K + k0 + 8 * kh);
    v8h b1 = *(const v8h*)(W + (col0 + n) * K + k0 + 16 + 8 * kh);
    v16h bm = __builtin_shufflevector(b0, b1, 0,1,2,3,4,5,6,7,8,9,10,11,12,13,14,15);
    acc = __builtin_amdgcn_wmma_f32_16x16x32_f16(false, a, false, bm, (short)0, acc,
                                                 false, false);
  }
#pragma unroll
  for (int v = 0; v < 8; ++v)
    Y[(row0 + 8 * kh + v) * N + col0 + n] = acc[v] + bias[col0 + n];
}

__global__ void fc_stats_kernel(const float* __restrict__ Y, float* __restrict__ st, int N)
{
  int f = blockIdx.x * blockDim.x + threadIdx.x;
  if (f >= N) return;
  float s = 0.f, q = 0.f;
  for (int b = 0; b < BB; ++b) { float x = Y[b * N + f]; s += x; q += x * x; }
  st[f * 2] = s; st[f * 2 + 1] = q;
}

__global__ void fc_apply_kernel(float* __restrict__ Y, const float* __restrict__ st, int N)
{
  int t = blockIdx.x * blockDim.x + threadIdx.x;
  if (t >= BB * N) return;
  int f = t % N;
  float mean = st[f * 2] * (1.f / BB);
  float var  = st[f * 2 + 1] * (1.f / BB) - mean * mean;
  Y[t] = fmaxf((Y[t] - mean) * rsqrtf(var + EPSB), 0.f);
}

__global__ __launch_bounds__(256) void heads_kernel(
    const float* __restrict__ X, const float* __restrict__ lw, const float* __restrict__ lb,
    const float* __restrict__ cw, const float* __restrict__ cb, float* __restrict__ out)
{
  const int b = blockIdx.x, tid = threadIdx.x;
  __shared__ float r1[256], r2[256];
  float s1 = 0.f, s2 = 0.f;
  for (int i = tid; i < 1024; i += 256) {
    float x = X[(size_t)b * 1024 + i];
    s1 += x * lw[i];
    s2 += x * cw[i];
  }
  r1[tid] = s1; r2[tid] = s2;
  __syncthreads();
  for (int s = 128; s > 0; s >>= 1) {
    if (tid < s) { r1[tid] += r1[tid + s]; r2[tid] += r2[tid + s]; }
    __syncthreads();
  }
  if (tid == 0) {
    out[b]      = r1[0] + lb[0];
    out[BB + b] = 1.f / (1.f + expf(-(r2[0] + cb[0])));
  }
}

// ---------------------------------------------------------------------------
// Host orchestration
// ---------------------------------------------------------------------------
extern "C" void kernel_launch(void* const* d_in, const int* in_sizes, int n_in,
                              void* d_out, int out_size, void* d_ws, size_t ws_size,
                              hipStream_t stream)
{
  (void)in_sizes; (void)n_in; (void)out_size; (void)ws_size;

  const float* pc   = (const float*)d_in[0];
  const float* gpc  = (const float*)d_in[1];
  const float* fc1b = (const float*)d_in[12];
  const float* fc2b = (const float*)d_in[14];

  char* p = (char*)d_ws;
  auto alloc = [&](size_t bytes) -> void* {
    void* r = (void*)p;
    p += (bytes + 255) & ~(size_t)255;
    return r;
  };

  float* xyz    = (float*)alloc((size_t)BB * NPTS1 * 3 * 4);
  int*   fidx1  = (int*)  alloc((size_t)BB * S1 * 4);
  float* nx1    = (float*)alloc((size_t)BB * S1 * 3 * 4);
  int*   gidx1  = (int*)  alloc((size_t)BB * S1 * NS1 * 4);
  int*   fidx2  = (int*)  alloc((size_t)BB * S2 * 4);
  float* nx2    = (float*)alloc((size_t)BB * S2 * 3 * 4);
  int*   gidx2  = (int*)  alloc((size_t)BB * S2 * NS2 * 4);
  float* feats1 = (float*)alloc((size_t)BB * 128 * S1 * 4);
  float* feats2 = (float*)alloc((size_t)BB * 256 * S2 * 4);
  float* sa3o   = (float*)alloc((size_t)BB * 512 * 4);
  float* fc1o   = (float*)alloc((size_t)BB * 1024 * 4);
  float* fc2o   = (float*)alloc((size_t)BB * 1024 * 4);
  float* stats  = (float*)alloc((size_t)7680 * 4);

  // f16 weights, K padded to multiple of 32
  _Float16* w_sa1_0 = (_Float16*)alloc((size_t)(64*32 + 64*64 + 128*64 +
                                               128*160 + 128*128 + 256*128 +
                                               256*288 + 256*256 + 512*256 +
                                               1024*512 + 1024*1024) * 2);
  _Float16* w_sa1_1 = w_sa1_0 + 64*32;
  _Float16* w_sa1_2 = w_sa1_1 + 64*64;
  _Float16* w_sa2_0 = w_sa1_2 + 128*64;
  _Float16* w_sa2_1 = w_sa2_0 + 128*160;
  _Float16* w_sa2_2 = w_sa2_1 + 128*128;
  _Float16* w_sa3_0 = w_sa2_2 + 256*128;
  _Float16* w_sa3_1 = w_sa3_0 + 256*288;
  _Float16* w_sa3_2 = w_sa3_1 + 256*256;
  _Float16* w_fc1   = w_sa3_2 + 512*256;
  _Float16* w_fc2   = w_fc1   + 1024*512;

  // stats layout (floats): sa1 @0 (512), sa2 @512 (1024), sa3 @1536 (2048),
  // fc1 @3584 (2048), fc2 @5632 (2048)
  float* st_sa1 = stats;
  float* st_sa2 = stats + 512;
  float* st_sa3 = stats + 1536;
  float* st_fc1 = stats + 3584;
  float* st_fc2 = stats + 5632;

  hipMemsetAsync(stats, 0, 7680 * 4, stream);

  auto cvt = [&](const void* src, _Float16* dst, int Co, int Ci, int Cip) {
    int n = Co * Cip;
    cvt_w_kernel<<<(n + 255) / 256, 256, 0, stream>>>((const float*)src, dst, Co, Ci, Cip);
  };
  cvt(d_in[2],  w_sa1_0,   64,    7,   32);
  cvt(d_in[3],  w_sa1_1,   64,   64,   64);
  cvt(d_in[4],  w_sa1_2,  128,   64,   64);
  cvt(d_in[5],  w_sa2_0,  128,  131,  160);
  cvt(d_in[6],  w_sa2_1,  128,  128,  128);
  cvt(d_in[7],  w_sa2_2,  256,  128,  128);
  cvt(d_in[8],  w_sa3_0,  256,  259,  288);
  cvt(d_in[9],  w_sa3_1,  256,  256,  256);
  cvt(d_in[10], w_sa3_2,  512,  256,  256);
  cvt(d_in[11], w_fc1,   1024,  512,  512);
  cvt(d_in[13], w_fc2,   1024, 1024, 1024);

  {
    int n = BB * NPTS1 * 3;
    concat_xyz_kernel<<<(n + 255) / 256, 256, 0, stream>>>(pc, gpc, xyz);
  }

  // ---- SA1 ----
  fps_kernel<<<BB, 256, 0, stream>>>(xyz, NPTS1, S1, fidx1);
  gather_nx_kernel<<<(BB * S1 * 3 + 255) / 256, 256, 0, stream>>>(xyz, fidx1, nx1, NPTS1, S1);
  ballq_kernel<<<(BB * S1 + 255) / 256, 256, 0, stream>>>(xyz, nx1, NPTS1, S1, NS1,
                                                          0.02f * 0.02f, gidx1);
  for (int pass = 0; pass < 4; ++pass)
    sa1_pass_kernel<<<BB * S1, 128, 0, stream>>>(xyz, nx1, gidx1,
                                                 w_sa1_0, w_sa1_1, w_sa1_2,
                                                 st_sa1, feats1, pass);

  // ---- SA2 ----
  fps_kernel<<<BB, 256, 0, stream>>>(nx1, S1, S2, fidx2);
  gather_nx_kernel<<<(BB * S2 * 3 + 255) / 256, 256, 0, stream>>>(nx1, fidx2, nx2, S1, S2);
  ballq_kernel<<<(BB * S2 + 255) / 256, 256, 0, stream>>>(nx1, nx2, S1, S2, NS2,
                                                          0.04f * 0.04f, gidx2);
  for (int pass = 0; pass < 4; ++pass)
    sa2_pass_kernel<<<BB * S2, 256, 0, stream>>>(nx1, feats1, nx2, gidx2,
                                                 w_sa2_0, w_sa2_1, w_sa2_2,
                                                 st_sa2, feats2, pass);

  // ---- SA3 ----
  for (int pass = 0; pass < 4; ++pass)
    sa3_pass_kernel<<<BB, 256, 0, stream>>>(nx2, feats2,
                                            w_sa3_0, w_sa3_1, w_sa3_2,
                                            st_sa3, sa3o, pass);

  // ---- FC head ----
  fc_gemm_kernel<<<64, 256, 0, stream>>>(sa3o, w_fc1, fc1b, fc1o, 512, 1024);
  fc_stats_kernel<<<4, 256, 0, stream>>>(fc1o, st_fc1, 1024);
  fc_apply_kernel<<<(BB * 1024 + 255) / 256, 256, 0, stream>>>(fc1o, st_fc1, 1024);

  fc_gemm_kernel<<<64, 256, 0, stream>>>(fc1o, w_fc2, fc2b, fc2o, 1024, 1024);
  fc_stats_kernel<<<4, 256, 0, stream>>>(fc2o, st_fc2, 1024);
  fc_apply_kernel<<<(BB * 1024 + 255) / 256, 256, 0, stream>>>(fc2o, st_fc2, 1024);

  heads_kernel<<<BB, 256, 0, stream>>>(fc2o,
                                       (const float*)d_in[15], (const float*)d_in[16],
                                       (const float*)d_in[17], (const float*)d_in[18],
                                       (float*)d_out);
}
